// UniGCNII_77464030151241
// MI455X (gfx1250) — matrix operand order repaired and checked by reading, not compiled, not run on hardware
//
#include <hip/hip_runtime.h>
#include <hip/hip_bf16.h>
#include <cmath>

// Problem constants (match reference)
#define NV    100000
#define ME    20000
#define NNZC  300000
#define DIM   256
#define NLAY  4
#define ALPHA 0.1f
#define LAMDA 0.5f

typedef __attribute__((ext_vector_type(8)))  __bf16 v8bf;
typedef __attribute__((ext_vector_type(16))) __bf16 v16bf;
typedef __attribute__((ext_vector_type(8)))  float  v8f;

// ---------------------------------------------------------------------------
// helpers
// ---------------------------------------------------------------------------
__device__ __forceinline__ unsigned short f2bf_u16(float f) {
    unsigned u = __builtin_bit_cast(unsigned, f);
    unsigned r = u + 0x7FFFu + ((u >> 16) & 1u);        // round-to-nearest-even
    return (unsigned short)(r >> 16);
}

__device__ __forceinline__ float bf2f(unsigned short s) {
    return __builtin_bit_cast(float, (unsigned)s << 16);
}

__device__ __forceinline__ v16bf load_a_frag(const __bf16* p) {
    // A fragment (16x32): lane holds one row; K chunk {c..c+15}
    v8bf lo = *(const v8bf*)(p);
    v8bf hi = *(const v8bf*)(p + 8);
    return __builtin_shufflevector(lo, hi, 0,1,2,3,4,5,6,7,8,9,10,11,12,13,14,15);
}

// ---------------------------------------------------------------------------
// WMMA GEMM: acc[rows, DIM] = A_bf16[rows, DIM] @ W_bf16[DIM, DIM]^T
//   mode 0: v = relu(acc + bias[col])
//   mode 1: v = relu((1-beta)*resid[row,col] + beta*acc)
//   mode 2: v = acc + bias[col]
// Stores f32 to `out` if non-null, bf16 to `outBf` if non-null.
// Each wave computes a 32x32 output tile (2x2 WMMA register tile): 4 WMMAs
// per 8 b128 loads. Block = 8 waves = all 256 columns. Grid: (rows/32, 1).
// ---------------------------------------------------------------------------
__global__ __launch_bounds__(256)
void gemm_wmma_bf16(const unsigned short* __restrict__ Abf,
                    const unsigned short* __restrict__ Wbf,
                    const float* __restrict__ bias,
                    const float* __restrict__ resid,
                    float* __restrict__ out,
                    unsigned short* __restrict__ outBf,
                    int mode, float beta)
{
    const int lane  = threadIdx.x & 31;
    const int wave  = threadIdx.x >> 5;
    const int half  = lane >> 4;       // 0: lanes 0-15, 1: lanes 16-31
    const int l15   = lane & 15;

    const int rowBase = blockIdx.x * 32;
    const int colBase = wave * 32;

    // A rows for the two row tiles handled by this lane
    const __bf16* __restrict__ aRow0 =
        (const __bf16*)Abf + (size_t)(rowBase + l15) * DIM + half * 16;
    const __bf16* __restrict__ aRow1 = aRow0 + (size_t)16 * DIM;
    // W rows (output channels) for the two column tiles
    const __bf16* __restrict__ bRow0 =
        (const __bf16*)Wbf + (size_t)(colBase + l15) * DIM + half * 16;
    const __bf16* __restrict__ bRow1 = bRow0 + (size_t)16 * DIM;

    v8f acc00 = {}, acc01 = {}, acc10 = {}, acc11 = {};
#pragma unroll
    for (int kt = 0; kt < DIM; kt += 32) {
        v16bf a0 = load_a_frag(aRow0 + kt);
        v16bf a1 = load_a_frag(aRow1 + kt);
        v16bf b0 = *(const v16bf*)(bRow0 + kt);
        v16bf b1 = *(const v16bf*)(bRow1 + kt);

        acc00 = __builtin_amdgcn_wmma_f32_16x16x32_bf16(
                    false, a0, false, b0, (short)0, acc00, false, false);
        acc01 = __builtin_amdgcn_wmma_f32_16x16x32_bf16(
                    false, a0, false, b1, (short)0, acc01, false, false);
        acc10 = __builtin_amdgcn_wmma_f32_16x16x32_bf16(
                    false, a1, false, b0, (short)0, acc10, false, false);
        acc11 = __builtin_amdgcn_wmma_f32_16x16x32_bf16(
                    false, a1, false, b1, (short)0, acc11, false, false);
    }

    // D layout: VGPR r -> row = tileRow + r + half*8, col = tileCol + l15
    v8f accs[4] = { acc00, acc01, acc10, acc11 };
#pragma unroll
    for (int t = 0; t < 4; ++t) {
        const int tileRow = rowBase + (t >> 1) * 16;
        const int outCol  = colBase + (t & 1) * 16 + l15;
        const float bcol  = (mode != 1) ? bias[outCol] : 0.0f;
#pragma unroll
        for (int r = 0; r < 8; ++r) {
            const int outRow = tileRow + r + half * 8;
            const size_t idx = (size_t)outRow * DIM + outCol;
            float v = accs[t][r];
            if (mode == 0) {
                v = fmaxf(v + bcol, 0.0f);
            } else if (mode == 1) {
                v = fmaxf((1.0f - beta) * resid[idx] + beta * v, 0.0f);
            } else {
                v = v + bcol;
            }
            if (out)   out[idx] = v;
            if (outBf) outBf[idx] = f2bf_u16(v);
        }
    }
}

// ---------------------------------------------------------------------------
// memory-bound kernels
// ---------------------------------------------------------------------------
__global__ void zero_f32(float* __restrict__ p, long long n) {
    long long i = (long long)blockIdx.x * blockDim.x + threadIdx.x;
    const long long stride = (long long)gridDim.x * blockDim.x;
    for (; i < n; i += stride) p[i] = 0.0f;
}

// f32 -> bf16, 4 elements per thread (n must be a multiple of 4)
__global__ void cvt_f32_bf16(const float* __restrict__ src,
                             unsigned short* __restrict__ dst, long long n4)
{
    long long i = (long long)blockIdx.x * blockDim.x + threadIdx.x;
    const long long stride = (long long)gridDim.x * blockDim.x;
    for (; i < n4; i += stride) {
        float4 f = ((const float4*)src)[i];
        uint2 u;
        u.x = (unsigned)f2bf_u16(f.x) | ((unsigned)f2bf_u16(f.y) << 16);
        u.y = (unsigned)f2bf_u16(f.z) | ((unsigned)f2bf_u16(f.w) << 16);
        ((uint2*)dst)[i] = u;
    }
}

__global__ void edge_count(const int* __restrict__ edges, float* __restrict__ cnt, int nnz) {
    int i = blockIdx.x * blockDim.x + threadIdx.x;
    if (i < nnz) atomicAdd(&cnt[edges[i]], 1.0f);
}

// dst[dstIdx[nz], f] += bf2f(srcBf[srcIdx[nz], f])  (vertex -> hyperedge)
__global__ __launch_bounds__(DIM)
void scatter_add_bf(const unsigned short* __restrict__ srcBf,
                    const int* __restrict__ srcIdx,
                    const int* __restrict__ dstIdx,
                    float* __restrict__ dst)
{
    const int nz = blockIdx.x;
    const int f  = threadIdx.x;
    const int s  = srcIdx[nz];
    const int d  = dstIdx[nz];
    atomicAdd(&dst[(size_t)d * DIM + f], bf2f(srcBf[(size_t)s * DIM + f]));
}

// dst[dstIdx[nz], f] += src[srcIdx[nz], f]  (hyperedge -> vertex, f32 source)
__global__ __launch_bounds__(DIM)
void scatter_add_f32(const float* __restrict__ src,
                     const int* __restrict__ srcIdx,
                     const int* __restrict__ dstIdx,
                     float* __restrict__ dst)
{
    const int nz = blockIdx.x;
    const int f  = threadIdx.x;
    const int s  = srcIdx[nz];
    const int d  = dstIdx[nz];
    atomicAdd(&dst[(size_t)d * DIM + f], src[(size_t)s * DIM + f]);
}

// Xe[e,f] = Xe[e,f] / max(cnt[e],1) * degE[e]
__global__ __launch_bounds__(DIM)
void edge_scale(float* __restrict__ Xe, const float* __restrict__ cnt,
                const float* __restrict__ degE)
{
    const int e = blockIdx.x;
    const int f = threadIdx.x;
    const float c = fmaxf(cnt[e], 1.0f);
    const size_t idx = (size_t)e * DIM + f;
    Xe[idx] = Xe[idx] / c * degE[e];
}

// Xi = (1-ALPHA)*degV*Xi + ALPHA*X0 ; writes f32 (residual) and bf16 (GEMM A)
__global__ __launch_bounds__(DIM)
void vert_combine(float* __restrict__ Xi, unsigned short* __restrict__ XiBf,
                  const float* __restrict__ X0, const float* __restrict__ degV)
{
    const int n = blockIdx.x;
    const int f = threadIdx.x;
    const size_t idx = (size_t)n * DIM + f;
    const float v = (1.0f - ALPHA) * degV[n] * Xi[idx] + ALPHA * X0[idx];
    Xi[idx]   = v;
    XiBf[idx] = f2bf_u16(v);
}

// ---------------------------------------------------------------------------
// driver
// ---------------------------------------------------------------------------
extern "C" void kernel_launch(void* const* d_in, const int* in_sizes, int n_in,
                              void* d_out, int out_size, void* d_ws, size_t ws_size,
                              hipStream_t stream)
{
    const float* x      = (const float*)d_in[0];
    const int*   vertex = (const int*)  d_in[1];
    const int*   edges  = (const int*)  d_in[2];
    const float* degV   = (const float*)d_in[3];
    const float* degE   = (const float*)d_in[4];
    const float* W0     = (const float*)d_in[5];
    const float* b0     = (const float*)d_in[6];
    const float* Wl     = (const float*)d_in[7];
    const float* Wout   = (const float*)d_in[8];
    const float* bout   = (const float*)d_in[9];
    float* out = (float*)d_out;

    const size_t NF = (size_t)NV * DIM;   // 25.6M elements
    const size_t MF = (size_t)ME * DIM;   // 5.12M elements
    const size_t WF = (size_t)DIM * DIM;  // 65536 elements

    // workspace layout
    float* X0f  = (float*)d_ws;                 // [NV,DIM] f32 residual source
    float* Xif  = X0f + NF;                     // [NV,DIM] f32 Xi (beta residual)
    float* Xe   = Xif + NF;                     // [ME,DIM] f32 edge accumulator
    float* cnt  = Xe  + MF;                     // [ME]
    unsigned short* Xbf  = (unsigned short*)(cnt + ME);   // [NV,DIM] bf16 activation
    unsigned short* Xibf = Xbf + NF;                      // [NV,DIM] bf16 GEMM A
    unsigned short* Wb   = Xibf + NF;                     // [6*DIM*DIM] bf16 weights

    unsigned short* Wb0   = Wb;
    unsigned short* Wbl   = Wb + WF;            // 4 layers contiguous
    unsigned short* Wbout = Wb + 5 * WF;

    const dim3 gemmGrid(NV / 32, 1);            // (3125, 1): 32 rows x 256 cols/block
    const dim3 gemmBlock(256);

    // hyperedge incidence counts
    zero_f32<<<256, 256, 0, stream>>>(cnt, (long long)ME);
    edge_count<<<(NNZC + 255) / 256, 256, 0, stream>>>(edges, cnt, NNZC);

    // one-time bf16 conversions: weights + input x (x_bf staged in Xibf)
    cvt_f32_bf16<<<512, 256, 0, stream>>>(W0,   Wb0,   (long long)(WF / 4));
    cvt_f32_bf16<<<1024, 256, 0, stream>>>(Wl,  Wbl,   (long long)(4 * WF / 4));
    cvt_f32_bf16<<<512, 256, 0, stream>>>(Wout, Wbout, (long long)(WF / 4));
    cvt_f32_bf16<<<4096, 256, 0, stream>>>(x,   Xibf,  (long long)(NF / 4));

    // X0 = relu(x @ W0^T + b0): f32 for residual, bf16 for gather
    gemm_wmma_bf16<<<gemmGrid, gemmBlock, 0, stream>>>(
        Xibf, Wb0, b0, nullptr, X0f, Xbf, 0, 0.0f);

    for (int i = 0; i < NLAY; ++i) {
        const float beta = logf(LAMDA / (float)(i + 1) + 1.0f);

        // Xe = segment_sum(X[vertex], edges) / cnt * degE   (bf16 gather source)
        zero_f32<<<2048, 256, 0, stream>>>(Xe, (long long)MF);
        scatter_add_bf<<<NNZC, DIM, 0, stream>>>(Xbf, vertex, edges, Xe);
        edge_scale<<<ME, DIM, 0, stream>>>(Xe, cnt, degE);

        // Xi = (1-a)*degV*segment_sum(Xe[edges], vertex) + a*X0
        zero_f32<<<4096, 256, 0, stream>>>(Xif, (long long)NF);
        scatter_add_f32<<<NNZC, DIM, 0, stream>>>(Xe, edges, vertex, Xif);
        vert_combine<<<NV, DIM, 0, stream>>>(Xif, Xibf, X0f, degV);

        // X = relu((1-beta)*Xi + beta*(Xi @ Wl[i]^T)) -> bf16 only
        gemm_wmma_bf16<<<gemmGrid, gemmBlock, 0, stream>>>(
            Xibf, Wbl + (size_t)i * WF, nullptr, Xif, nullptr, Xbf, 1, beta);
    }

    // out = X @ Wout^T + bout  (f32 output)
    gemm_wmma_bf16<<<gemmGrid, gemmBlock, 0, stream>>>(
        Xbf, Wbout, bout, nullptr, out, nullptr, 2, 0.0f);
}
